// Block_71253507441369
// MI455X (gfx1250) — matrix operand order repaired and checked by reading, not compiled, not run on hardware
//
#include <hip/hip_runtime.h>

// ---------------- problem constants (from reference) ----------------
#define DD   256     // D
#define LL   1024    // L
#define NB   16      // B*C sequences
#define DI   512     // expand*D
#define MD_  2048    // C*D
#define M1   16384   // NB*LL rows of the sequence GEMMs

typedef __attribute__((ext_vector_type(16))) __bf16 v16bf;
typedef __attribute__((ext_vector_type(8)))  float  v8f;
typedef __attribute__((ext_vector_type(4)))  unsigned int u32x4;
typedef __attribute__((ext_vector_type(8)))  int          i32x8;
typedef __attribute__((ext_vector_type(4)))  int          i32x4;

#if defined(__HIP_DEVICE_COMPILE__) && __has_builtin(__builtin_amdgcn_tensor_load_to_lds) && __has_builtin(__builtin_amdgcn_s_wait_tensorcnt)
#define HAVE_TDM 1
#else
#define HAVE_TDM 0
#endif

struct __align__(16) U4 { unsigned x, y, z, w; };
union ABFrag { v16bf v; U4 q[2]; };

__device__ __forceinline__ unsigned short f2bf(float f) {
  unsigned u = __float_as_uint(f);
  unsigned r = u + 0x7FFFu + ((u >> 16) & 1u);  // round-to-nearest-even
  return (unsigned short)(r >> 16);
}
__device__ __forceinline__ float sigmoidf_(float x) { return 1.0f / (1.0f + __expf(-x)); }

// ---------------- generic f32 -> bf16 conversion ----------------
__global__ __launch_bounds__(256) void cvt_f32_bf16_k(const float* __restrict__ s,
                                                      unsigned short* __restrict__ d, int n) {
  int i = blockIdx.x * 256 + threadIdx.x;
  if (i < n) d[i] = f2bf(s[i]);
}

// ---------------- LN over D (first norm) -> bf16 u[(n*L+l)*D + d] ----------------
__global__ __launch_bounds__(256) void ln1_kernel(const float* __restrict__ x,
    const float* __restrict__ g, const float* __restrict__ b,
    unsigned short* __restrict__ u_bf) {
  __shared__ float s1[256], s2[256];
  int pos = blockIdx.x;              // n*L + l
  int n = pos >> 10, l = pos & 1023;
  int d = threadIdx.x;
  float v = x[((long long)(n * DD + d)) * LL + l];
  s1[d] = v; s2[d] = v * v; __syncthreads();
  for (int o = 128; o > 0; o >>= 1) {
    if (d < o) { s1[d] += s1[d + o]; s2[d] += s2[d + o]; }
    __syncthreads();
  }
  float mean = s1[0] * (1.0f / DD);
  float var  = s2[0] * (1.0f / DD) - mean * mean;
  float y = (v - mean) * rsqrtf(var + 1e-5f) * g[d] + b[d];
  u_bf[(long long)pos * DD + d] = f2bf(y);
}

// ---------------- residual + LN2: x2 = x + mo^T ; hh = LN(x2) (bf16) ----------------
__global__ __launch_bounds__(256) void res_ln2_kernel(const float* __restrict__ x,
    const float* __restrict__ mo, const float* __restrict__ g, const float* __restrict__ b,
    float* __restrict__ x2, unsigned short* __restrict__ hh_bf) {
  __shared__ float s1[256], s2[256];
  int pos = blockIdx.x;
  int n = pos >> 10, l = pos & 1023;
  int d = threadIdx.x;
  long long xi = ((long long)(n * DD + d)) * LL + l;   // flat (B,C,D,L) == (b, o, l)
  float v = x[xi] + mo[(long long)pos * DD + d];
  x2[xi] = v;
  s1[d] = v; s2[d] = v * v; __syncthreads();
  for (int o = 128; o > 0; o >>= 1) {
    if (d < o) { s1[d] += s1[d + o]; s2[d] += s2[d + o]; }
    __syncthreads();
  }
  float mean = s1[0] * (1.0f / DD);
  float var  = s2[0] * (1.0f / DD) - mean * mean;
  float y = (v - mean) * rsqrtf(var + 1e-5f) * g[d] + b[d];
  hh_bf[xi] = f2bf(y);
}

// ---------------- causal depthwise conv (k=4) + SiLU ----------------
__global__ __launch_bounds__(256) void conv_silu_kernel(const float* __restrict__ xz,
    const float* __restrict__ cw, const float* __restrict__ cb,
    float* __restrict__ xi_f, unsigned short* __restrict__ xi_bf) {
  long long idx = (long long)blockIdx.x * 256 + threadIdx.x;   // M1*DI
  int ch = (int)(idx & (DI - 1));
  long long ml = idx >> 9;                 // n*L + l
  int l = (int)(ml & (LL - 1));
  float a = cb[ch];
  #pragma unroll
  for (int t = 0; t < 4; ++t) {
    int lt = l - 3 + t;
    if (lt >= 0) a += cw[ch * 4 + t] * xz[(ml - (3 - t)) * 1024 + ch];
  }
  float s = a * sigmoidf_(a);
  xi_f[ml * DI + ch]  = s;
  xi_bf[ml * DI + ch] = f2bf(s);
}

// ---------------- dt = softplus(x_dbl[:, :16] @ dt_w^T + dt_b) ----------------
__global__ __launch_bounds__(256) void dt_kernel(const float* __restrict__ x_dbl,
    const float* __restrict__ dtw, const float* __restrict__ dtb, float* __restrict__ dt) {
  long long idx = (long long)blockIdx.x * 256 + threadIdx.x;   // M1*DI
  int ch = (int)(idx & (DI - 1));
  long long m = idx >> 9;
  const float* xr = x_dbl + m * 48;
  float a = dtb[ch];
  #pragma unroll
  for (int r = 0; r < 16; ++r) a += xr[r] * dtw[ch * 16 + r];
  dt[m * DI + ch] = (a > 20.0f) ? a : log1pf(__expf(a));
}

// ---------------- selective scan + D-skip + SiLU(z) gate -> bf16 y ----------------
__global__ __launch_bounds__(256) void scan_kernel(const float* __restrict__ xi_f,
    const float* __restrict__ dt, const float* __restrict__ x_dbl,
    const float* __restrict__ A_log, const float* __restrict__ D_skip,
    const float* __restrict__ xz, unsigned short* __restrict__ y_bf) {
  int n  = blockIdx.x;
  int ch = blockIdx.y * 256 + threadIdx.x;
  float A[16], h[16];
  #pragma unroll
  for (int s = 0; s < 16; ++s) { A[s] = -__expf(A_log[ch * 16 + s]); h[s] = 0.0f; }
  float dsk = D_skip[ch];
  for (int l = 0; l < LL; ++l) {
    long long m = (long long)n * LL + l;
    float dtv = dt[m * DI + ch];
    float uv  = xi_f[m * DI + ch];
    const float* bc = x_dbl + m * 48;
    float du = dtv * uv;
    float y = 0.0f;
    #pragma unroll
    for (int s = 0; s < 16; ++s) {
      h[s] = h[s] * __expf(dtv * A[s]) + du * bc[16 + s];
      y += h[s] * bc[32 + s];
    }
    y += uv * dsk;
    float z = xz[m * 1024 + DI + ch];
    y *= z * sigmoidf_(z);
    y_bf[m * DI + ch] = f2bf(y);
  }
}

// ---------------- bf16 WMMA GEMM: C(MxN) = A(MxK) * B ----------------
// B_IS_NK: B stored as N x K row-major (transposed weight). else K x N row-major.
// EPI: 0 = f32 out ; 1 = +bias, ReLU, bf16 out ; 2 = +bias +resid, f32 out.
#define BM 128
#define BN 128
#define BK 32
#define LDSP 40   // BK + 8 pad (80B row stride keeps 16B alignment)

#if HAVE_TDM
// TDM 2D tile load: rows x BK bf16 tile, row stride ld_elems, into LDS at lds_off with
// 16B pad every 64B (reproduces the LDSP=40 layout). OOB rows/cols are zero-filled by HW.
__device__ __forceinline__ void tdm_load_2d(unsigned lds_off, const void* gptr,
                                            unsigned rows_rem, unsigned cols_rem,
                                            unsigned ld_elems) {
  unsigned long long ga = (unsigned long long)gptr;
  u32x4 g0;
  g0[0] = 1u;                                      // count = 1 (valid user descriptor)
  g0[1] = lds_off;                                 // LDS byte address
  g0[2] = (unsigned)ga;                            // global_addr[31:0]
  g0[3] = (unsigned)(ga >> 32) | (2u << 30);       // global_addr[56:32] | type=2
  i32x8 g1;
  g1[0] = (1 << 16)                                // data_size = 2 bytes
        | (1 << 20)                                // pad_enable
        | (3 << 22)                                // pad_interval: 16 DWORDs (64B)
        | (3 << 25);                               // pad_amount: 4 DWORDs (16B)
  g1[1] = (int)((cols_rem & 0xFFFFu) << 16);       // tensor_dim0[15:0]
  g1[2] = (int)((cols_rem >> 16) | ((rows_rem & 0xFFFFu) << 16)); // dim0 hi | dim1 lo
  g1[3] = (int)((rows_rem >> 16) | ((unsigned)BK << 16));         // dim1 hi | tile_dim0
  g1[4] = BM;                                      // tile_dim1 (tile_dim2 = 0)
  g1[5] = (int)ld_elems;                           // tensor_dim0_stride[31:0]
  g1[6] = 0;                                       // stride0 hi | stride1 lo
  g1[7] = 0;                                       // stride1 hi
  i32x4 z4 = {0, 0, 0, 0};
  i32x8 z8 = {0, 0, 0, 0, 0, 0, 0, 0};
  // 6-arg toolchain form: (group0, group1, group2, group3, extra group, cpol)
  __builtin_amdgcn_tensor_load_to_lds(g0, g1, z4, z4, z8, 0);
}
#endif

template<bool B_IS_NK, int EPI>
__global__ __launch_bounds__(256) void gemm_bf16_k(
    const unsigned short* __restrict__ A, int lda,
    const unsigned short* __restrict__ Bsrc, int ldb, long long strB,
    float* __restrict__ outF, unsigned short* __restrict__ outBF, int ldc, long long strOut,
    const float* __restrict__ bias,
    const float* __restrict__ resid, long long strRes,
    int M, int Nn, int K) {
  __shared__ __align__(16) unsigned short sA[BM * LDSP];
  __shared__ __align__(16) unsigned short sB[BN * LDSP];

  const int tid  = threadIdx.x;
  const int m0   = blockIdx.x * BM;
  const int n0   = blockIdx.y * BN;
  const unsigned short* Bp = Bsrc + (long long)blockIdx.z * strB;

  const int wave = tid >> 5, lane = tid & 31;
  const int wm = wave & 3, wn = wave >> 2;            // 4x2 wave grid, 32x64 per wave
  const int lrow = lane & 15;
  const int cA = (lane < 16) ? 0 : 8;                 // A frag: K in {cA..cA+7, cA+16..cA+23}
  const int kB = (lane < 16) ? 0 : 8;                 // B frag: K in {kB..kB+7, kB+16..kB+23}

  v8f acc[2][4];
  #pragma unroll
  for (int i = 0; i < 2; ++i)
    #pragma unroll
    for (int j = 0; j < 4; ++j) {
      v8f z = {0.f,0.f,0.f,0.f,0.f,0.f,0.f,0.f};
      acc[i][j] = z;
    }

  for (int k0 = 0; k0 < K; k0 += BK) {
#if HAVE_TDM
    // --- TDM staging: wave 0 issues 2D tile DMAs, tracked by TENSORcnt ---
    if (tid < 32) {
      tdm_load_2d((unsigned)(unsigned long long)(const void*)sA,
                  A + (long long)m0 * lda + k0, (unsigned)(M - m0), (unsigned)(K - k0),
                  (unsigned)lda);
      if (B_IS_NK)
        tdm_load_2d((unsigned)(unsigned long long)(const void*)sB,
                    Bp + (long long)n0 * ldb + k0, (unsigned)(Nn - n0), (unsigned)(K - k0),
                    (unsigned)ldb);
      __builtin_amdgcn_s_wait_tensorcnt(0);
    }
    if (!B_IS_NK) {
      for (int i = tid; i < BK * (BN / 8); i += 256) {
        int k = i >> 4, n8 = (i & 15) << 3;
        union { U4 v; unsigned short s[8]; } tv;
        int gn = n0 + n8;
        if (gn + 7 < Nn) {
          tv.v = *(const U4*)(Bp + (long long)(k0 + k) * ldb + gn);
        } else {
          #pragma unroll
          for (int j = 0; j < 8; ++j)
            tv.s[j] = (gn + j < Nn) ? Bp[(long long)(k0 + k) * ldb + gn + j] : (unsigned short)0;
        }
        #pragma unroll
        for (int j = 0; j < 8; ++j) sB[(n8 + j) * LDSP + k] = tv.s[j];
      }
    }
#else
    if (k0 + BK < K) {                                 // pull next K-tile toward L2/WGP$
      __builtin_prefetch((const void*)(A + (long long)m0 * lda + (k0 + BK)), 0, 1);
      if (B_IS_NK)
        __builtin_prefetch((const void*)(Bp + (long long)n0 * ldb + (k0 + BK)), 0, 1);
      else
        __builtin_prefetch((const void*)(Bp + (long long)(k0 + BK) * ldb + n0), 0, 1);
    }
    // stage A tile: row-major [m][k]
    for (int i = tid; i < BM * BK / 8; i += 256) {
      int r = i >> 2, c8 = (i & 3) << 3;
      U4 v; v.x = v.y = v.z = v.w = 0u;
      int gr = m0 + r;
      if (gr < M) v = *(const U4*)(A + (long long)gr * lda + k0 + c8);
      *(U4*)(&sA[r * LDSP + c8]) = v;
    }
    // stage B tile: column-of-K layout [n][k]
    if (B_IS_NK) {
      for (int i = tid; i < BN * BK / 8; i += 256) {
        int nr = i >> 2, c8 = (i & 3) << 3;
        U4 v; v.x = v.y = v.z = v.w = 0u;
        int gn = n0 + nr;
        if (gn < Nn) v = *(const U4*)(Bp + (long long)gn * ldb + k0 + c8);
        *(U4*)(&sB[nr * LDSP + c8]) = v;
      }
    } else {
      for (int i = tid; i < BK * (BN / 8); i += 256) {
        int k = i >> 4, n8 = (i & 15) << 3;
        union { U4 v; unsigned short s[8]; } tv;
        int gn = n0 + n8;
        if (gn + 7 < Nn) {
          tv.v = *(const U4*)(Bp + (long long)(k0 + k) * ldb + gn);
        } else {
          #pragma unroll
          for (int j = 0; j < 8; ++j)
            tv.s[j] = (gn + j < Nn) ? Bp[(long long)(k0 + k) * ldb + gn + j] : (unsigned short)0;
        }
        #pragma unroll
        for (int j = 0; j < 8; ++j) sB[(n8 + j) * LDSP + k] = tv.s[j];
      }
    }
#endif
    __syncthreads();

    ABFrag af[2], bfr[4];
    #pragma unroll
    for (int t = 0; t < 2; ++t) {
      const unsigned short* p = &sA[(wm * 32 + t * 16 + lrow) * LDSP + cA];
      af[t].q[0] = *(const U4*)(p);
      af[t].q[1] = *(const U4*)(p + 16);
    }
    #pragma unroll
    for (int t = 0; t < 4; ++t) {
      const unsigned short* p = &sB[(wn * 64 + t * 16 + lrow) * LDSP + kB];
      bfr[t].q[0] = *(const U4*)(p);
      bfr[t].q[1] = *(const U4*)(p + 16);
    }
    #pragma unroll
    for (int i = 0; i < 2; ++i)
      #pragma unroll
      for (int j = 0; j < 4; ++j)
        acc[i][j] = __builtin_amdgcn_wmma_f32_16x16x32_bf16(
            false, af[i].v, false, bfr[j].v, (short)0, acc[i][j], false, false);
    __syncthreads();
  }

  // epilogue: C/D layout — VGPR r: M = base + (lane<16 ? r : 8+r), N = lane&15
  float* of = outF ? (outF + (long long)blockIdx.z * strOut) : nullptr;
  unsigned short* ob = outBF ? (outBF + (long long)blockIdx.z * strOut) : nullptr;
  const float* rs = resid ? (resid + (long long)blockIdx.z * strRes) : nullptr;
  #pragma unroll
  for (int i = 0; i < 2; ++i) {
    int mbase = m0 + wm * 32 + i * 16 + ((lane < 16) ? 0 : 8);
    #pragma unroll
    for (int j = 0; j < 4; ++j) {
      int col = n0 + wn * 64 + j * 16 + lrow;
      if (col < Nn) {
        #pragma unroll
        for (int r = 0; r < 8; ++r) {
          int row = mbase + r;
          if (row < M) {
            float v = acc[i][j][r];
            long long idx = (long long)row * ldc + col;
            if (EPI == 0) {
              of[idx] = v;
            } else if (EPI == 1) {
              v += bias[row];
              ob[idx] = f2bf(v > 0.0f ? v : 0.0f);
            } else {
              of[idx] = v + bias[row] + rs[idx];
            }
          }
        }
      }
    }
  }
}

// ---------------- host-side orchestration ----------------
extern "C" void kernel_launch(void* const* d_in, const int* in_sizes, int n_in,
                              void* d_out, int out_size, void* d_ws, size_t ws_size,
                              hipStream_t stream) {
  (void)in_sizes; (void)n_in; (void)out_size; (void)ws_size;
  const float* x         = (const float*)d_in[0];
  const float* n1_g      = (const float*)d_in[1];
  const float* n1_b      = (const float*)d_in[2];
  const float* n2_g      = (const float*)d_in[3];
  const float* n2_b      = (const float*)d_in[4];
  const float* in_proj_w = (const float*)d_in[5];
  const float* conv_w    = (const float*)d_in[6];
  const float* conv_b    = (const float*)d_in[7];
  const float* x_proj_w  = (const float*)d_in[8];
  const float* dt_proj_w = (const float*)d_in[9];
  const float* dt_proj_b = (const float*)d_in[10];
  const float* A_log     = (const float*)d_in[11];
  const float* D_skip    = (const float*)d_in[12];
  const float* out_proj_w= (const float*)d_in[13];
  const float* pw1_w     = (const float*)d_in[14];
  const float* pw1_b     = (const float*)d_in[15];
  const float* pw2_w     = (const float*)d_in[16];
  const float* pw2_b     = (const float*)d_in[17];
  float* out = (float*)d_out;

  char* ws = (char*)d_ws;
  size_t off = 0;
  auto alloc = [&](size_t bytes) -> void* {
    off = (off + 255) & ~(size_t)255;
    void* p = ws + off; off += bytes; return p;
  };
  unsigned short* u_bf   = (unsigned short*)alloc((size_t)M1 * DD * 2);
  unsigned short* w_in   = (unsigned short*)alloc((size_t)1024 * DD * 2);
  unsigned short* w_xp   = (unsigned short*)alloc((size_t)48 * DI * 2);
  unsigned short* w_op   = (unsigned short*)alloc((size_t)DD * DI * 2);
  unsigned short* w_p1   = (unsigned short*)alloc((size_t)2 * MD_ * MD_ * 2);
  unsigned short* w_p2   = (unsigned short*)alloc((size_t)MD_ * 2 * MD_ * 2);
  float*          xz     = (float*)alloc((size_t)M1 * 1024 * 4);
  float*          xi_f   = (float*)alloc((size_t)M1 * DI * 4);
  unsigned short* xi_bf  = (unsigned short*)alloc((size_t)M1 * DI * 2);
  float*          x_dbl  = (float*)alloc((size_t)M1 * 48 * 4);
  float*          dt     = (float*)alloc((size_t)M1 * DI * 4);
  unsigned short* y_bf   = (unsigned short*)alloc((size_t)M1 * DI * 2);
  float*          mo     = (float*)alloc((size_t)M1 * DD * 4);
  float*          x2     = (float*)alloc((size_t)2 * MD_ * LL * 4);
  unsigned short* hh_bf  = (unsigned short*)alloc((size_t)2 * MD_ * LL * 2);
  unsigned short* mid_bf = (unsigned short*)alloc((size_t)2 * 2 * MD_ * LL * 2);

  auto cvt = [&](const float* s, unsigned short* d, int n) {
    cvt_f32_bf16_k<<<dim3((n + 255) / 256), dim3(256), 0, stream>>>(s, d, n);
  };
  cvt(in_proj_w, w_in, 1024 * DD);
  cvt(x_proj_w,  w_xp, 48 * DI);
  cvt(out_proj_w,w_op, DD * DI);
  cvt(pw1_w,     w_p1, 2 * MD_ * MD_);
  cvt(pw2_w,     w_p2, MD_ * 2 * MD_);

  // 1) LN1 -> u (bf16)
  ln1_kernel<<<dim3(M1), dim3(256), 0, stream>>>(x, n1_g, n1_b, u_bf);

  // 2) in_proj: xz[M1,1024] = u[M1,256] * W(1024x256)^T
  gemm_bf16_k<true, 0><<<dim3(M1 / BM, 1024 / BN, 1), dim3(256), 0, stream>>>(
      u_bf, DD, w_in, DD, 0, xz, nullptr, 1024, 0, nullptr, nullptr, 0, M1, 1024, DD);

  // 3) causal dwconv + SiLU
  conv_silu_kernel<<<dim3((M1 * DI) / 256), dim3(256), 0, stream>>>(xz, conv_w, conv_b, xi_f, xi_bf);

  // 4) x_proj: x_dbl[M1,48] = xi[M1,512] * W(48x512)^T  (TDM zero-fills rows 48..127)
  gemm_bf16_k<true, 0><<<dim3(M1 / BM, 1, 1), dim3(256), 0, stream>>>(
      xi_bf, DI, w_xp, DI, 0, x_dbl, nullptr, 48, 0, nullptr, nullptr, 0, M1, 48, DI);

  // 5) dt = softplus(rank-16 proj)
  dt_kernel<<<dim3((M1 * DI) / 256), dim3(256), 0, stream>>>(x_dbl, dt_proj_w, dt_proj_b, dt);

  // 6) selective scan + gate -> y (bf16)
  scan_kernel<<<dim3(NB, DI / 256), dim3(256), 0, stream>>>(xi_f, dt, x_dbl, A_log, D_skip, xz, y_bf);

  // 7) out_proj: mo[M1,256] = y[M1,512] * W(256x512)^T
  gemm_bf16_k<true, 0><<<dim3(M1 / BM, DD / BN, 1), dim3(256), 0, stream>>>(
      y_bf, DI, w_op, DI, 0, mo, nullptr, DD, 0, nullptr, nullptr, 0, M1, DD, DI);

  // 8) residual + LN2 -> x2 (f32), hh (bf16)
  res_ln2_kernel<<<dim3(M1), dim3(256), 0, stream>>>(x, mo, n2_g, n2_b, x2, hh_bf);

  // 9) pw1: mid[b][4096,1024] = relu(W1(4096x2048) * hh[b](2048x1024) + b1), bf16 out
  gemm_bf16_k<false, 1><<<dim3((2 * MD_) / BM, LL / BN, 2), dim3(256), 0, stream>>>(
      w_p1, MD_, hh_bf, LL, (long long)MD_ * LL,
      nullptr, mid_bf, LL, (long long)2 * MD_ * LL,
      pw1_b, nullptr, 0, 2 * MD_, LL, MD_);

  // 10) pw2 + bias + residual -> d_out
  gemm_bf16_k<false, 2><<<dim3(MD_ / BM, LL / BN, 2), dim3(256), 0, stream>>>(
      w_p2, 2 * MD_, mid_bf, LL, (long long)2 * MD_ * LL,
      out, nullptr, LL, (long long)MD_ * LL,
      pw2_b, x2, (long long)MD_ * LL, MD_, LL, 2 * MD_);
}
